// ECLGCNN_43087111913921
// MI455X (gfx1250) — matrix operand order repaired and checked by reading, not compiled, not run on hardware
//
#include <hip/hip_runtime.h>

// ---------------- Model constants ----------------
#define BB   1024
#define TT   48
#define NPGc 32
#define FF   5
#define HH   512
#define EPGc 256
#define NTOT 1572864LL          // B*T*NPG
#define ETOT 12582912LL         // B*T*EPG
#define NG   49152              // B*T graphs
#define G4   2048               // 4*H
#define KC   672                // 160 + 512 (concat K for gates GEMM)

typedef _Float16 v8h  __attribute__((ext_vector_type(8)));
typedef _Float16 v16h __attribute__((ext_vector_type(16)));
typedef float    v8f  __attribute__((ext_vector_type(8)));

static __device__ __forceinline__ float sigf(float x) {
    return 1.0f / (1.0f + __expf(-x));
}

// ============================================================
// Kernel 1: fused ChebConv (K=3) per 32-node graph.
// One wave per graph; degrees/norm/SpMM fully in LDS (ds_add_f32).
// ============================================================
__global__ __launch_bounds__(256) void k_cheb(const float* __restrict__ x,
                                              const float* __restrict__ ea,
                                              const int*   __restrict__ ei,
                                              const float* __restrict__ W,   // [3][5][5]
                                              const float* __restrict__ bias,// [5]
                                              float* __restrict__ y) {
    __shared__ float sX[8][NPGc][FF];
    __shared__ float sDeg[8][NPGc];
    __shared__ float sDinv[8][NPGc];
    __shared__ float sT1[8][NPGc][FF];
    __shared__ float sT2[8][NPGc][FF];

    const int lane = threadIdx.x & 31;
    const int w    = threadIdx.x >> 5;
    const int g    = blockIdx.x * 8 + w;
    const int nodeBase = g * NPGc;

    float xr[FF];
#pragma unroll
    for (int f = 0; f < FF; ++f) {
        xr[f] = x[(size_t)(nodeBase + lane) * FF + f];
        sX[w][lane][f]  = xr[f];
        sT1[w][lane][f] = 0.f;
        sT2[w][lane][f] = 0.f;
    }
    sDeg[w][lane] = 0.f;
    __syncthreads();

    int   sl[8], dl[8];
    float wt[8];
    const size_t ebase = (size_t)g * EPGc;
#pragma unroll
    for (int j = 0; j < 8; ++j) {
        size_t e = ebase + (size_t)j * 32 + lane;
        sl[j] = ei[e] - nodeBase;                 // src local
        dl[j] = ei[(size_t)ETOT + e] - nodeBase;  // dst local
        wt[j] = ea[e];
        atomicAdd(&sDeg[w][sl[j]], wt[j]);        // deg = segsum over src
    }
    __syncthreads();

    float dg = sDeg[w][lane];
    sDinv[w][lane] = (dg > 0.f) ? rsqrtf(dg) : 0.f;
    __syncthreads();

    float nrm[8];
#pragma unroll
    for (int j = 0; j < 8; ++j) {
        nrm[j] = -sDinv[w][sl[j]] * wt[j] * sDinv[w][dl[j]];
#pragma unroll
        for (int f = 0; f < FF; ++f)
            atomicAdd(&sT1[w][dl[j]][f], nrm[j] * sX[w][sl[j]][f]); // T1 = L~ x
    }
    __syncthreads();

#pragma unroll
    for (int j = 0; j < 8; ++j) {
#pragma unroll
        for (int f = 0; f < FF; ++f)
            atomicAdd(&sT2[w][dl[j]][f], 2.f * nrm[j] * sT1[w][sl[j]][f]);
    }
    __syncthreads();

    float t1v[FF], t2v[FF];
#pragma unroll
    for (int f = 0; f < FF; ++f) {
        t1v[f] = sT1[w][lane][f];
        t2v[f] = sT2[w][lane][f] - xr[f];        // T2 = 2 L~ T1 - T0
    }
#pragma unroll
    for (int f = 0; f < FF; ++f) {
        float o = bias[f];
#pragma unroll
        for (int j = 0; j < FF; ++j) {
            o += xr[j]  * W[ 0 + j * FF + f];
            o += t1v[j] * W[25 + j * FF + f];
            o += t2v[j] * W[50 + j * FF + f];
        }
        y[(size_t)(nodeBase + lane) * FF + f] = o;
    }
}

// ============================================================
// Kernel 2: per-sample BatchNorm (training stats) + sigmoid -> f16
// ============================================================
__global__ __launch_bounds__(256) void k_bn(const float* __restrict__ y,
                                            const float* __restrict__ gamma,
                                            const float* __restrict__ beta,
                                            _Float16* __restrict__ xo) {
    __shared__ float red[256][10];
    const int b = blockIdx.x, tid = threadIdx.x;
    const float* yb = y + (size_t)b * 1536 * FF;

    float s[FF] = {0, 0, 0, 0, 0}, q[FF] = {0, 0, 0, 0, 0};
    for (int r = tid; r < 1536; r += 256) {
#pragma unroll
        for (int f = 0; f < FF; ++f) {
            float v = yb[r * FF + f];
            s[f] += v; q[f] += v * v;
        }
    }
#pragma unroll
    for (int f = 0; f < FF; ++f) { red[tid][f] = s[f]; red[tid][5 + f] = q[f]; }
    for (int st = 128; st > 0; st >>= 1) {
        __syncthreads();
        if (tid < st)
#pragma unroll
            for (int u = 0; u < 10; ++u) red[tid][u] += red[tid + st][u];
    }
    __syncthreads();

    float scale[FF], shift[FF];
#pragma unroll
    for (int f = 0; f < FF; ++f) {
        float mean = red[0][f] * (1.f / 1536.f);
        float var  = red[0][5 + f] * (1.f / 1536.f) - mean * mean;
        float sc   = gamma[f] * rsqrtf(var + 1e-5f);
        scale[f] = sc;
        shift[f] = beta[f] - mean * sc;
    }
    _Float16* xb = xo + (size_t)b * 1536 * FF;
    for (int r = tid; r < 1536; r += 256) {
#pragma unroll
        for (int f = 0; f < FF; ++f) {
            float v = yb[r * FF + f] * scale[f] + shift[f];
            xb[r * FF + f] = (_Float16)sigf(v);
        }
    }
}

// ============================================================
// Kernel 3: build Wcat^T [2048 x 672] in f16  (cols: [W_ih | W_hh])
// ============================================================
__global__ void k_prepw(const float* __restrict__ Wih, const float* __restrict__ Whh,
                        _Float16* __restrict__ wc, int total) {
    int i = blockIdx.x * 256 + threadIdx.x;
    if (i >= total) return;
    int n = i / KC, k = i % KC;
    float v = (k < 160) ? Wih[n * 160 + k] : Whh[n * HH + (k - 160)];
    wc[i] = (_Float16)v;
}

// ============================================================
// Kernel 4: zero h_in (f16) and c (f32)
// ============================================================
__global__ void k_init(float* __restrict__ c, _Float16* __restrict__ hf) {
    int idx = blockIdx.x * 256 + threadIdx.x;
    if (idx < BB * HH) { c[idx] = 0.f; hf[idx] = (_Float16)0.f; }
}

// ============================================================
// Kernel 5: fused LSTM step t (GEMM + cell):
//   gates = [x_t | h_in] (f16, K=672) @ Wcat (f16), f32 WMMA acc.
// Gate-grouped N tiling: a block owns the SAME n-slice of all 4 gates
// (cols n, 512+n, 1024+n, 1536+n), so i/f/g/o of an output element sit
// in one lane's accumulators -> cell update is pure register math.
// Block = 256 thr (8 waves), block tile 64(M) x 64(n); wave 16(M) x 32(n) x 4 gates.
// h is ping-ponged (read h_in, write h_out) to avoid intra-step races;
// c is exclusively owned per element -> updated in place.
// ============================================================
__global__ __launch_bounds__(256) void k_step(const _Float16* __restrict__ xf,
                                              const _Float16* __restrict__ hin,
                                              const _Float16* __restrict__ wc,
                                              const float* __restrict__ bih,
                                              const float* __restrict__ bhh,
                                              float* __restrict__ c,
                                              _Float16* __restrict__ hout,
                                              float* __restrict__ hs, int t) {
    const int tid  = threadIdx.x;
    const int lane = tid & 31, wave = tid >> 5;
    const int mb = blockIdx.x & 15;      // 16 M-blocks of 64 rows
    const int nb = blockIdx.x >> 4;      // 8 n-blocks of 64 hidden units
    const int m0 = mb * 64 + (wave & 3) * 16;
    const int n0 = nb * 64 + (wave >> 2) * 32;
    const int row  = m0 + (lane & 15);
    const int half = lane >> 4;

    v8f acc[2][4];                       // [n-subtile][gate]
#pragma unroll
    for (int s = 0; s < 2; ++s)
#pragma unroll
        for (int g = 0; g < 4; ++g) acc[s][g] = v8f{0};

    const _Float16* xrow = xf + ((size_t)row * TT + t) * 160;
    const _Float16* hrow = hin + (size_t)row * HH;

    for (int kt = 0; kt < 21; ++kt) {
        const int k0 = kt * 32;
        // A tile (16x32): lane<16 holds K{0..15}, lane>=16 K{16..31} (matches B)
        const _Float16* ap = (k0 < 160) ? (xrow + k0) : (hrow + (k0 - 160));
        v16h av = *(const v16h*)(ap + half * 16);
        __builtin_prefetch(ap + 32, 0, 0);  // next K tile -> global_prefetch_b8
#pragma unroll
        for (int s = 0; s < 2; ++s) {
#pragma unroll
            for (int g = 0; g < 4; ++g) {
                // B tile (32x16): lane = column (n of gate g), 16 contiguous K per lane
                const int col = g * HH + n0 + s * 16 + (lane & 15);
                const _Float16* bp = wc + (size_t)col * KC + k0 + half * 16;
                v16h bv = *(const v16h*)bp;
                acc[s][g] = __builtin_amdgcn_wmma_f32_16x16x32_f16(
                    false, av, false, bv, (short)0, acc[s][g], false, false);
            }
        }
    }

    // Epilogue: LSTM cell (torch gate order i,f,g,o) on registers.
    // C/D layout: VGPR v -> M = v + 8*half, N = lane%16
#pragma unroll
    for (int s = 0; s < 2; ++s) {
        const int nn = n0 + s * 16 + (lane & 15);
        const float bi = bih[nn]          + bhh[nn];
        const float bf = bih[HH + nn]     + bhh[HH + nn];
        const float bg = bih[2 * HH + nn] + bhh[2 * HH + nn];
        const float bo = bih[3 * HH + nn] + bhh[3 * HH + nn];
#pragma unroll
        for (int v = 0; v < 8; ++v) {
            const int r = m0 + v + half * 8;
            const size_t idx = (size_t)r * HH + nn;
            const float iv = sigf(acc[s][0][v] + bi);
            const float fv = sigf(acc[s][1][v] + bf);
            const float gv = tanhf(acc[s][2][v] + bg);
            const float ov = sigf(acc[s][3][v] + bo);
            const float cn = fv * c[idx] + iv * gv;
            c[idx] = cn;
            const float hv = ov * tanhf(cn);
            hout[idx] = (_Float16)hv;
            hs[((size_t)r * TT + t) * HH + nn] = sigf(hv);
        }
    }
}

// ============================================================
// Kernel 6: final linear head: out[b,o] = sig( sum_j hs_sig[b,j]*lin_W[o,j] + lb[o] )
// ============================================================
__global__ __launch_bounds__(256) void k_head(const float* __restrict__ hs,
                                              const float* __restrict__ lw,
                                              const float* __restrict__ lb,
                                              float* __restrict__ out) {
    __shared__ float red[256][4];
    const int b = blockIdx.x, tid = threadIdx.x;
    const float* hb = hs + (size_t)b * (TT * HH);
    float p[4] = {0, 0, 0, 0};
    for (int j = tid; j < TT * HH; j += 256) {
        float v = hb[j];
#pragma unroll
        for (int o = 0; o < 4; ++o) p[o] += v * lw[(size_t)o * (TT * HH) + j];
    }
#pragma unroll
    for (int o = 0; o < 4; ++o) red[tid][o] = p[o];
    for (int st = 128; st > 0; st >>= 1) {
        __syncthreads();
        if (tid < st)
#pragma unroll
            for (int o = 0; o < 4; ++o) red[tid][o] += red[tid + st][o];
    }
    if (tid == 0)
#pragma unroll
        for (int o = 0; o < 4; ++o) out[b * 4 + o] = sigf(red[0][o] + lb[o]);
}

// ============================================================
extern "C" void kernel_launch(void* const* d_in, const int* in_sizes, int n_in,
                              void* d_out, int out_size, void* d_ws, size_t ws_size,
                              hipStream_t stream) {
    (void)in_sizes; (void)n_in; (void)out_size; (void)ws_size;
    const float* x     = (const float*)d_in[0];
    const float* ea    = (const float*)d_in[1];
    const float* chW   = (const float*)d_in[2];
    const float* chB   = (const float*)d_in[3];
    const float* gamma = (const float*)d_in[4];
    const float* beta  = (const float*)d_in[5];
    const float* Wih   = (const float*)d_in[6];
    const float* Whh   = (const float*)d_in[7];
    const float* bih   = (const float*)d_in[8];
    const float* bhh   = (const float*)d_in[9];
    const float* lw    = (const float*)d_in[10];
    const float* lb    = (const float*)d_in[11];
    const int*   ei    = (const int*)d_in[12];
    float* out = (float*)d_out;

    char* w = (char*)d_ws;
    size_t off = 0;
    auto alloc = [&](size_t bytes) -> void* {
        void* p = w + off;
        off = (off + bytes + 255) & ~(size_t)255;
        return p;
    };
    float*    y_cheb = (float*)   alloc((size_t)NTOT * FF * 4);        // 31.5 MB
    _Float16* xf     = (_Float16*)alloc((size_t)BB * TT * 160 * 2);    // 15.7 MB
    _Float16* hbuf0  = (_Float16*)alloc((size_t)BB * HH * 2);          //  1.0 MB
    _Float16* hbuf1  = (_Float16*)alloc((size_t)BB * HH * 2);          //  1.0 MB
    _Float16* wc     = (_Float16*)alloc((size_t)G4 * KC * 2);          //  2.75 MB
    float*    cbuf   = (float*)   alloc((size_t)BB * HH * 4);          //  2.1 MB
    float*    hs     = (float*)   alloc((size_t)BB * TT * HH * 4);     // 100.7 MB

    k_cheb <<<NG / 8, 256, 0, stream>>>(x, ea, ei, chW, chB, y_cheb);
    k_bn   <<<BB,     256, 0, stream>>>(y_cheb, gamma, beta, xf);
    k_prepw<<<(G4 * KC + 255) / 256, 256, 0, stream>>>(Wih, Whh, wc, G4 * KC);
    k_init <<<(BB * HH + 255) / 256, 256, 0, stream>>>(cbuf, hbuf0);

    for (int t = 0; t < TT; ++t) {
        _Float16* hin  = (t & 1) ? hbuf1 : hbuf0;
        _Float16* hout = (t & 1) ? hbuf0 : hbuf1;
        k_step<<<128, 256, 0, stream>>>(xf, hin, wc, bih, bhh, cbuf, hout, hs, t);
    }
    k_head<<<BB, 256, 0, stream>>>(hs, lw, lb, out);
}
